// HNet_89352499626294
// MI455X (gfx1250) — matrix operand order repaired and checked by reading, not compiled, not run on hardware
//
#include <hip/hip_runtime.h>
#include <hip/hip_bf16.h>

typedef __attribute__((ext_vector_type(16))) _Float16 v16h;
typedef __attribute__((ext_vector_type(8)))  _Float16 v8h;
typedef __attribute__((ext_vector_type(8)))  float    v8f;

#define BB   4
#define LL   8192
#define DD   512
#define NSEG 64
#define SEGLEN (LL / NSEG)   // 128
#define APITCH 520           // LDS row pitch in halfs (bank-conflict-free, 16B aligned)
#define MROWS  32            // positions per workgroup (two 16-row WMMA tiles)

union FragH { v16h v; v8h h[2]; };

__device__ __forceinline__ v8f wmma_f16(const FragH& a, const FragH& b, v8f c) {
    return __builtin_amdgcn_wmma_f32_16x16x32_f16(
        false, a.v, false, b.v, (short)0, c, false, false);
}

// ---------------------------------------------------------------------------
// Kernel 0: convert f32 weights -> f16 workspace (row-major, same layout).
// Also leaves the 1 MB of f16 weights hot in L2 for the GEMM kernel.
// ---------------------------------------------------------------------------
__global__ void wcvt_kernel(const float* __restrict__ qw,
                            const float* __restrict__ kw,
                            _Float16* __restrict__ wq_h,
                            _Float16* __restrict__ wk_h) {
    const int n = DD * DD;
    int idx = blockIdx.x * blockDim.x + threadIdx.x;
    if (idx < n) {
        wq_h[idx] = (_Float16)qw[idx];
        wk_h[idx] = (_Float16)kw[idx];
    }
}

// ---------------------------------------------------------------------------
// Kernel 1 (WMMA): fused q=Wq*h[l], k=Wk*h[l+1], cos_sim, prob.
//   grid = (L/32, B), block = 256 (8 waves). M = 32 rows per WG (tiles T0,T1)
//   so every B fragment feeds 2 A tiles (4 WMMAs per bq/bk pair).
//   Wave w owns output dims [w*64, w*64+64) as four 16-wide N tiles,
//   processed in two groups of 2 to bound register pressure.
//   Only sum(q^2), sum(k^2), sum(q*k) per row survive.
// ---------------------------------------------------------------------------
__global__ void __launch_bounds__(256)
cosprob_kernel(const float* __restrict__ hidden,
               const _Float16* __restrict__ wq,
               const _Float16* __restrict__ wk,
               float* __restrict__ prob) {
    __shared__ _Float16 shh[(MROWS + 1) * APITCH];   // 33 rows
    __shared__ float red[MROWS * 3];

    const int b  = blockIdx.y;
    const int l0 = blockIdx.x * MROWS;        // positions l0..l0+31 (valid if <= L-2)
    const int tid  = threadIdx.x;
    const int wave = tid >> 5;
    const int lane = tid & 31;
    const int m    = lane & 15;
    const int hi   = lane >> 4;

    // Stage 33 hidden rows (l0 .. l0+32, clamped) into LDS as f16.
    for (int idx = tid; idx < (MROWS + 1) * DD; idx += 256) {
        int r = idx >> 9, c = idx & (DD - 1);
        int row = l0 + r; if (row > LL - 1) row = LL - 1;
        shh[r * APITCH + c] = (_Float16)hidden[((size_t)b * LL + row) * DD + c];
    }
    if (tid < MROWS * 3) red[tid] = 0.0f;
    __syncthreads();

    v8f accQ[2][4] = {};
    v8f accK[2][4] = {};

    const _Float16* rq0 = &shh[m * APITCH];            // T0 q rows: l0+m
    const _Float16* rk0 = &shh[(m + 1) * APITCH];      // T0 k rows: l0+m+1
    const _Float16* rq1 = &shh[(m + 16) * APITCH];     // T1 q rows: l0+16+m
    const _Float16* rk1 = &shh[(m + 17) * APITCH];     // T1 k rows: l0+16+m+1

    for (int k0 = 0; k0 < DD; k0 += 32) {
        // A fragments (16x32 f16): halfs[0..7]=K hi*8.., halfs[8..15]=K 16+hi*8..
        FragH aq0, ak0, aq1, ak1;
        aq0.h[0] = *(const v8h*)(rq0 + k0 + hi * 8);
        aq0.h[1] = *(const v8h*)(rq0 + k0 + 16 + hi * 8);
        ak0.h[0] = *(const v8h*)(rk0 + k0 + hi * 8);
        ak0.h[1] = *(const v8h*)(rk0 + k0 + 16 + hi * 8);
        aq1.h[0] = *(const v8h*)(rq1 + k0 + hi * 8);
        aq1.h[1] = *(const v8h*)(rq1 + k0 + 16 + hi * 8);
        ak1.h[0] = *(const v8h*)(rk1 + k0 + hi * 8);
        ak1.h[1] = *(const v8h*)(rk1 + k0 + 16 + hi * 8);

        const int koff = k0 + hi * 8;    // B: halfs[0..7]=K hi*8.., halfs[8..15]=K 16+hi*8..

#pragma unroll
        for (int g = 0; g < 2; ++g) {    // two groups of 2 N tiles
            const int nt0 = g * 2;
            const int nA = wave * 64 + nt0 * 16 + m;
            const int nB = nA + 16;

            // 4 distinct B fragments -> loads batch into one clause,
            // one staged wait, then 8 back-to-back WMMAs.
            FragH bqA, bqB, bkA, bkB;
            const _Float16* pqA = wq + (size_t)nA * DD + koff;
            const _Float16* pqB = wq + (size_t)nB * DD + koff;
            const _Float16* pkA = wk + (size_t)nA * DD + koff;
            const _Float16* pkB = wk + (size_t)nB * DD + koff;
            bqA.h[0] = *(const v8h*)(pqA);  bqA.h[1] = *(const v8h*)(pqA + 16);
            bqB.h[0] = *(const v8h*)(pqB);  bqB.h[1] = *(const v8h*)(pqB + 16);
            bkA.h[0] = *(const v8h*)(pkA);  bkA.h[1] = *(const v8h*)(pkA + 16);
            bkB.h[0] = *(const v8h*)(pkB);  bkB.h[1] = *(const v8h*)(pkB + 16);

            accQ[0][nt0]     = wmma_f16(aq0, bqA, accQ[0][nt0]);
            accQ[1][nt0]     = wmma_f16(aq1, bqA, accQ[1][nt0]);
            accQ[0][nt0 + 1] = wmma_f16(aq0, bqB, accQ[0][nt0 + 1]);
            accQ[1][nt0 + 1] = wmma_f16(aq1, bqB, accQ[1][nt0 + 1]);
            accK[0][nt0]     = wmma_f16(ak0, bkA, accK[0][nt0]);
            accK[1][nt0]     = wmma_f16(ak1, bkA, accK[1][nt0]);
            accK[0][nt0 + 1] = wmma_f16(ak0, bkB, accK[0][nt0 + 1]);
            accK[1][nt0 + 1] = wmma_f16(ak1, bkB, accK[1][nt0 + 1]);
        }
    }

    // Per-lane partial sums over this wave's 64 N columns, per row tile.
#pragma unroll
    for (int t = 0; t < 2; ++t) {
        float sq[8] = {}, sk[8] = {}, sqk[8] = {};
#pragma unroll
        for (int nt = 0; nt < 4; ++nt)
#pragma unroll
            for (int j = 0; j < 8; ++j) {
                float q = accQ[t][nt][j], k = accK[t][nt][j];
                sq[j]  += q * q;
                sk[j]  += k * k;
                sqk[j] += q * k;
            }
        // Reduce over N within each 16-lane half (C/D layout: lanes 0-15 N=0..15).
#pragma unroll
        for (int j = 0; j < 8; ++j)
#pragma unroll
            for (int off = 1; off < 16; off <<= 1) {
                sq[j]  += __shfl_xor(sq[j],  off, 32);
                sk[j]  += __shfl_xor(sk[j],  off, 32);
                sqk[j] += __shfl_xor(sqk[j], off, 32);
            }
        if (m == 0) {   // lanes 0 (rows 0..7) and 16 (rows 8..15)
            const int rbase = t * 16 + (hi ? 8 : 0);
#pragma unroll
            for (int j = 0; j < 8; ++j) {
                atomicAdd(&red[(rbase + j) * 3 + 0], sq[j]);
                atomicAdd(&red[(rbase + j) * 3 + 1], sk[j]);
                atomicAdd(&red[(rbase + j) * 3 + 2], sqk[j]);
            }
        }
    }
    __syncthreads();

    if (tid < MROWS) {
        const int l = l0 + tid;
        if (l <= LL - 2) {
            float a = red[tid * 3 + 0];
            float c = red[tid * 3 + 1];
            float d = red[tid * 3 + 2];
            float nq = fmaxf(sqrtf(a), 1e-12f);
            float nk = fmaxf(sqrtf(c), 1e-12f);
            float cosv = d / (nq * nk);
            float p = 0.5f * (1.0f - cosv);
            p = fminf(fmaxf(p, 0.0f), 1.0f);
            prob[(size_t)b * LL + l + 1] = p;
        }
        if (l0 == 0 && tid == 0) prob[(size_t)b * LL] = 1.0f;   // state=None path
    }
}

// ---------------------------------------------------------------------------
// The sort/scan/gather tail collapses to a gated EMA stream:
//   if p[l] > 0.5: state = (1-p)*state + p*h[l];  out[l] = residual[l] + state
// Segmented scan, 3 phases.
// ---------------------------------------------------------------------------

// Phase 1: per-segment local EMA (state from 0) + scalar decay product.
__global__ void __launch_bounds__(512)
seg_local_kernel(const float* __restrict__ hidden,
                 const float* __restrict__ prob,
                 float* __restrict__ Aseg,   // (B, NSEG)
                 float* __restrict__ Bseg) { // (B, NSEG, D)
    __shared__ float pp[SEGLEN];
    const int b = blockIdx.y, s = blockIdx.x, ch = threadIdx.x;
    const int l0 = s * SEGLEN;
    if (ch < SEGLEN) pp[ch] = prob[(size_t)b * LL + l0 + ch];
    __syncthreads();

    float local = 0.0f, aprod = 1.0f;
    for (int i = 0; i < SEGLEN; ++i) {
        float p = pp[i];
        if (p > 0.5f) {                       // block-uniform branch
            float h = hidden[((size_t)b * LL + l0 + i) * DD + ch];
            local = (1.0f - p) * local + p * h;
            aprod *= (1.0f - p);
        }
    }
    Bseg[((size_t)b * NSEG + s) * DD + ch] = local;
    if (ch == 0) Aseg[b * NSEG + s] = aprod;
}

// Phase 2: sequential combine across segments (in place: Bseg becomes the
// prefix state *entering* each segment). state seeded with det_state.
__global__ void __launch_bounds__(512)
seg_combine_kernel(const float* __restrict__ det,
                   const float* __restrict__ Aseg,
                   float* __restrict__ Bseg) {
    const int b = blockIdx.x, ch = threadIdx.x;
    float state = det[b * DD + ch];
    for (int s = 0; s < NSEG; ++s) {
        const size_t idx = ((size_t)b * NSEG + s) * DD + ch;
        float bv = Bseg[idx];
        float a  = Aseg[b * NSEG + s];
        Bseg[idx] = state;                    // prefix state entering segment s
        state = a * state + bv;
    }
}

// Phase 3: replay each segment with correct prefix state, add residual.
__global__ void __launch_bounds__(512)
seg_final_kernel(const float* __restrict__ hidden,
                 const float* __restrict__ residual,
                 const float* __restrict__ prob,
                 const float* __restrict__ Pseg,   // prefix states (was Bseg)
                 float* __restrict__ out) {
    __shared__ float pp[SEGLEN];
    const int b = blockIdx.y, s = blockIdx.x, ch = threadIdx.x;
    const int l0 = s * SEGLEN;
    if (ch < SEGLEN) pp[ch] = prob[(size_t)b * LL + l0 + ch];
    __syncthreads();

    float state = Pseg[((size_t)b * NSEG + s) * DD + ch];
    for (int i = 0; i < SEGLEN; ++i) {
        const size_t idx = ((size_t)b * LL + l0 + i) * DD + ch;
        float p = pp[i];
        if (p > 0.5f)
            state = (1.0f - p) * state + p * hidden[idx];
        out[idx] = residual[idx] + state;     // STE coef == 1 in forward
    }
}

// ---------------------------------------------------------------------------
extern "C" void kernel_launch(void* const* d_in, const int* in_sizes, int n_in,
                              void* d_out, int out_size, void* d_ws, size_t ws_size,
                              hipStream_t stream) {
    const float* hidden   = (const float*)d_in[0];
    const float* residual = (const float*)d_in[1];
    const float* qw       = (const float*)d_in[2];
    const float* kw       = (const float*)d_in[3];
    const float* det      = (const float*)d_in[4];
    float* out = (float*)d_out;

    char* ws = (char*)d_ws;
    constexpr size_t WQ_OFF   = 0;
    constexpr size_t WK_OFF   = WQ_OFF + (size_t)DD * DD * sizeof(_Float16);
    constexpr size_t PROB_OFF = WK_OFF + (size_t)DD * DD * sizeof(_Float16);
    constexpr size_t ASEG_OFF = PROB_OFF + (size_t)BB * LL * sizeof(float);
    constexpr size_t BSEG_OFF = ASEG_OFF + 4096;
    _Float16* wq_h = (_Float16*)(ws + WQ_OFF);
    _Float16* wk_h = (_Float16*)(ws + WK_OFF);
    float* prob = (float*)(ws + PROB_OFF);
    float* Aseg = (float*)(ws + ASEG_OFF);
    float* Bseg = (float*)(ws + BSEG_OFF);

    // 0) weights f32 -> f16
    wcvt_kernel<<<(DD * DD + 255) / 256, 256, 0, stream>>>(qw, kw, wq_h, wk_h);

    // 1) fused WMMA GEMM + cosine router probabilities
    cosprob_kernel<<<dim3(LL / MROWS, BB), 256, 0, stream>>>(hidden, wq_h, wk_h, prob);

    // 2) gated-EMA segmented scan (3 phases)
    seg_local_kernel<<<dim3(NSEG, BB), DD, 0, stream>>>(hidden, prob, Aseg, Bseg);
    seg_combine_kernel<<<BB, DD, 0, stream>>>(det, Aseg, Bseg);
    seg_final_kernel<<<dim3(NSEG, BB), DD, 0, stream>>>(hidden, residual, prob, Bseg, out);
}
